// Multi_Head_Attention_28372553957593
// MI455X (gfx1250) — matrix operand order repaired and checked by reading, not compiled, not run on hardware
//
#include <hip/hip_runtime.h>
#include <hip/hip_bf16.h>

// ---------------------------------------------------------------------------
// Types for CDNA5 WMMA (wave32)
// ---------------------------------------------------------------------------
typedef __bf16 bf16x8 __attribute__((ext_vector_type(8)));
typedef __bf16 v16bf  __attribute__((ext_vector_type(16)));
typedef float  v8f    __attribute__((ext_vector_type(8)));

union FragAB { v16bf v; bf16x8 h[2]; };

__device__ __forceinline__ v8f zero8() {
    v8f z = {0.f, 0.f, 0.f, 0.f, 0.f, 0.f, 0.f, 0.f};
    return z;
}

// round-to-nearest-even f32 -> bf16 (bit pattern as ushort)
__device__ __forceinline__ unsigned short f2bf(float f) {
    union { float f; unsigned u; } v; v.f = f;
    unsigned r = v.u + 0x7FFFu + ((v.u >> 16) & 1u);
    return (unsigned short)(r >> 16);
}

// ---------------------------------------------------------------------------
// CDNA5 async global->LDS copy (16B per lane), tracked by ASYNCcnt.
// ---------------------------------------------------------------------------
__device__ __forceinline__ void async_ld16(void* lds_ptr, const void* gptr) {
    unsigned lds_off = (unsigned)(uintptr_t)lds_ptr;
    unsigned long long ga = (unsigned long long)(uintptr_t)gptr;
    asm volatile("global_load_async_to_lds_b128 %0, %1, off"
                 :: "v"(lds_off), "v"(ga) : "memory");
}
__device__ __forceinline__ void wait_async0() {
    asm volatile("s_wait_asynccnt 0x0" ::: "memory");
}
__device__ __forceinline__ void wait_async6() {
    asm volatile("s_wait_asynccnt 0x6" ::: "memory");
}

// ---------------------------------------------------------------------------
// Kernel 1: f32 -> bf16 convert, 8 elems/thread
// ---------------------------------------------------------------------------
__global__ __launch_bounds__(256) void conv_kernel(
    const float* __restrict__ X, unsigned short* __restrict__ Xb, size_t n8)
{
    size_t i = (size_t)blockIdx.x * blockDim.x + threadIdx.x;
    if (i >= n8) return;
    const float4* xv = reinterpret_cast<const float4*>(X) + i * 2;
    float4 a = xv[0], b = xv[1];
    unsigned short o[8];
    o[0] = f2bf(a.x); o[1] = f2bf(a.y); o[2] = f2bf(a.z); o[3] = f2bf(a.w);
    o[4] = f2bf(b.x); o[5] = f2bf(b.y); o[6] = f2bf(b.z); o[7] = f2bf(b.w);
    *(reinterpret_cast<uint4*>(Xb) + i) = *reinterpret_cast<const uint4*>(o);
}

// ---------------------------------------------------------------------------
// Kernel 2: f32 W[K][N] -> bf16 Wt[N][K], LDS-tiled 32x32 transpose
// ---------------------------------------------------------------------------
__global__ __launch_bounds__(256) void conv_tr_kernel(
    const float* __restrict__ W, unsigned short* __restrict__ Wt, int Kd, int Nd)
{
    __shared__ float tile[32][33];
    const int kb = blockIdx.y * 32, nb = blockIdx.x * 32;
    const int tx = threadIdx.x & 31, ty = threadIdx.x >> 5;  // 32 x 8
#pragma unroll
    for (int i = 0; i < 4; i++)
        tile[ty + i * 8][tx] = W[(size_t)(kb + ty + i * 8) * Nd + nb + tx];
    __syncthreads();
#pragma unroll
    for (int i = 0; i < 4; i++)
        Wt[(size_t)(nb + ty + i * 8) * Kd + kb + tx] = f2bf(tile[tx][ty + i * 8]);
}

// ---------------------------------------------------------------------------
// Kernel 2b: bf16 V[(b,t)][(h,d)] -> Vt[(b,h,d)][t]  (per-head transpose)
// 32x32 LDS tiles; both global accesses coalesced.
// ---------------------------------------------------------------------------
__global__ __launch_bounds__(256) void vtr_kernel(
    const unsigned short* __restrict__ Vin,  // [8192][2048]
    unsigned short* __restrict__ Vt)         // [(b*16+h)*128+d][2048]
{
    __shared__ unsigned short tile[32][33];
    const int mb = blockIdx.x * 32;          // (b,t) rows
    const int nb = blockIdx.y * 32;          // (h,d) cols (32 | 128, no h split)
    const int tx = threadIdx.x & 31, ty = threadIdx.x >> 5;  // 32 x 8
#pragma unroll
    for (int i = 0; i < 4; i++)
        tile[ty + i * 8][tx] = Vin[(size_t)(mb + ty + i * 8) * 2048 + nb + tx];
    __syncthreads();
    const int b  = mb >> 11;
    const int t0 = mb & 2047;
#pragma unroll
    for (int i = 0; i < 4; i++) {
        int n = nb + ty + i * 8;
        int h = n >> 7, d = n & 127;
        size_t orow = (size_t)(b * 16 + h) * 128 + d;
        Vt[orow * 2048 + t0 + tx] = tile[tx][ty + i * 8];
    }
}

// ---------------------------------------------------------------------------
// Kernel 3: GEMM  C[8192][2048] = A[8192][2048] @ Bt[2048][2048]^T
// bf16 in, f32 accum. 256 threads (8 waves), 128x256 C tile, K-tile 32,
// double-buffered LDS fed by global_load_async_to_lds_b128 with a 2-deep
// prefetch pipeline and a fully peeled, branch-free steady-state loop.
//   OUTF32 == 0 : write bf16 C ; OUTF32 == 1 : write f32 C + bias[n]
// ---------------------------------------------------------------------------
#define TILE_K 32
#define LDK (TILE_K + 8)

template <int OUTF32>
__global__ __launch_bounds__(256) void gemm_bf16_kernel(
    const unsigned short* __restrict__ A,
    const unsigned short* __restrict__ Bt,
    unsigned short* __restrict__ Cbf,
    float* __restrict__ Cf,
    const float* __restrict__ bias)
{
    constexpr int K = 2048;
    constexpr int N = 2048;
    constexpr int NIT = K / TILE_K;  // 64 K-steps

    __shared__ alignas(16) unsigned short As[2][128][LDK];
    __shared__ alignas(16) unsigned short Bs[2][256][LDK];

    const int tid  = threadIdx.x;
    const int lane = tid & 31;
    const int wave = tid >> 5;
    const int half = lane >> 4;
    const int l16  = lane & 15;
    const int wr   = wave & 1;    // 64-row group
    const int wc   = wave >> 1;   // 64-col group (0..3)
    const int m0 = blockIdx.y * 128;
    const int n0 = blockIdx.x * 256;

    v8f acc[4][4];
#pragma unroll
    for (int i = 0; i < 4; i++)
#pragma unroll
        for (int j = 0; j < 4; j++) acc[i][j] = zero8();

    // issue one K-tile's async loads: A 512 chunks + B 1024 chunks, 6/thread
    auto issue_tile = [&](int k0, int buf) {
#pragma unroll
        for (int i = 0; i < 2; i++) {
            int c = tid + i * 256;
            int r = c >> 2, cc = (c & 3) * 8;
            async_ld16(&As[buf][r][cc], &A[(size_t)(m0 + r) * K + k0 + cc]);
        }
#pragma unroll
        for (int i = 0; i < 4; i++) {
            int c = tid + i * 256;
            int r = c >> 2, cc = (c & 3) * 8;
            async_ld16(&Bs[buf][r][cc], &Bt[(size_t)(n0 + r) * K + k0 + cc]);
        }
    };

    // one K-step of WMMA work from LDS buffer `buf`
    auto compute = [&](int buf) {
        FragAB fa[4], fb[4];
#pragma unroll
        for (int tn = 0; tn < 4; tn++) {
            const unsigned short* p = &Bs[buf][wc * 64 + tn * 16 + l16][half * 8];
            fb[tn].h[0] = *reinterpret_cast<const bf16x8*>(p);
            fb[tn].h[1] = *reinterpret_cast<const bf16x8*>(p + 16);
        }
#pragma unroll
        for (int tm = 0; tm < 4; tm++) {
            const unsigned short* p = &As[buf][wr * 64 + tm * 16 + l16][half * 8];
            fa[tm].h[0] = *reinterpret_cast<const bf16x8*>(p);
            fa[tm].h[1] = *reinterpret_cast<const bf16x8*>(p + 16);
        }
#pragma unroll
        for (int tm = 0; tm < 4; tm++)
#pragma unroll
            for (int tn = 0; tn < 4; tn++)
                acc[tm][tn] = __builtin_amdgcn_wmma_f32_16x16x32_bf16(
                    false, fa[tm].v, false, fb[tn].v, (short)0, acc[tm][tn],
                    false, false);
    };

    // 2-deep prefetch pipeline, steady state is branch-free
    issue_tile(0, 0);
    issue_tile(TILE_K, 1);
    for (int it = 0; it < NIT - 2; it += 2) {
        wait_async6();            // tile it resident (in-order completion)
        __syncthreads();
        compute(0);
        __syncthreads();
        issue_tile((it + 2) * TILE_K, 0);
        wait_async6();            // tile it+1 resident
        __syncthreads();
        compute(1);
        __syncthreads();
        issue_tile((it + 3) * TILE_K, 1);
    }
    wait_async6();
    __syncthreads();
    compute(0);                   // tile NIT-2
    __syncthreads();
    wait_async0();
    compute(1);                   // tile NIT-1

    // epilogue: lane holds (m = r + half*8, n = l16) per accumulator VGPR r
#pragma unroll
    for (int tm = 0; tm < 4; tm++) {
#pragma unroll
        for (int tn = 0; tn < 4; tn++) {
            int rowb = m0 + wr * 64 + tm * 16 + half * 8;
            int col  = n0 + wc * 64 + tn * 16 + l16;
#pragma unroll
            for (int r = 0; r < 8; r++) {
                float vv = acc[tm][tn][r];
                size_t idx = (size_t)(rowb + r) * N + col;
                if (OUTF32) Cf[idx] = vv + bias[col];
                else        Cbf[idx] = f2bf(vv);
            }
        }
    }
}

// ---------------------------------------------------------------------------
// Kernel 4: causal flash attention, one block per (b, h, 64-row q tile).
// Q/K/Vt tiles all staged with global_load_async_to_lds_b128 (straight-line,
// fully unrolled issue sequences).
// Vt layout: [(b*16+h)*128+d][t] so its LDS tile is k-contiguous per d-row.
// ---------------------------------------------------------------------------
__global__ __launch_bounds__(256) void attn_kernel(
    const unsigned short* __restrict__ Qg,
    const unsigned short* __restrict__ Kg,
    const unsigned short* __restrict__ Vtg,
    unsigned short* __restrict__ Og)
{
    const int T = 2048, D = 2048, HD = 128;
    __shared__ alignas(16) unsigned short Qs[64][HD + 8];
    __shared__ alignas(16) unsigned short Ks[32][HD + 8];
    __shared__ alignas(16) unsigned short Vts[HD][32 + 8];  // [d][k]
    __shared__ float Ss[64][36];
    __shared__ alignas(16) unsigned short Ps[64][40];
    __shared__ float m_s[64], l_s[64], al_s[64];

    const int qt = blockIdx.x, h = blockIdx.y, b = blockIdx.z;
    const int row0 = qt * 64;
    const int tid  = threadIdx.x;
    const int lane = tid & 31, wave = tid >> 5;
    const int half = lane >> 4, l16 = lane & 15;
    const int qt16 = wave & 3;   // 16-row strip of the 64-row q tile
    const int dgrp = wave >> 2;  // 64-col half of hd

    const size_t baseQ  = (size_t)(b * T + row0) * D + h * HD;
    const size_t baseK  = (size_t)(b * T) * D + h * HD;
    const size_t baseVt = (size_t)(b * 16 + h) * HD * T;   // rows of length T

    // stage Q tile (64x128 bf16): 1024 chunks, 4/thread, straight-line
#pragma unroll
    for (int i = 0; i < 4; i++) {
        int c = tid + i * 256;
        int r = c >> 4, cc = (c & 15) * 8;
        async_ld16(&Qs[r][cc], &Qg[baseQ + (size_t)r * D + cc]);
    }
    if (tid < 64) { m_s[tid] = -3.0e38f; l_s[tid] = 0.f; }

    v8f oacc[4];
#pragma unroll
    for (int j = 0; j < 4; j++) oacc[j] = zero8();

    wait_async0();
    __syncthreads();

    const int ktmax = 2 * qt + 1;  // cover k <= row0+63
    for (int kt = 0; kt <= ktmax; kt++) {
        const int k0 = kt * 32;
        // K tile (32 x 128): 512 chunks, 2/thread
#pragma unroll
        for (int i = 0; i < 2; i++) {
            int c = tid + i * 256;
            int r = c >> 4, cc = (c & 15) * 8;
            async_ld16(&Ks[r][cc], &Kg[baseK + (size_t)(k0 + r) * D + cc]);
        }
        // Vt tile (128 d-rows x 32 k): 512 chunks, 2/thread
#pragma unroll
        for (int i = 0; i < 2; i++) {
            int c = tid + i * 256;
            int r = c >> 2, cc = (c & 3) * 8;
            async_ld16(&Vts[r][cc], &Vtg[baseVt + (size_t)r * T + k0 + cc]);
        }
        wait_async0();
        __syncthreads();

        // S = Q @ K^T : wave -> tile (qt16, dgrp), reduce over d = 4 x 32
        {
            v8f sacc = zero8();
#pragma unroll
            for (int kc = 0; kc < 4; kc++) {
                FragAB fa, fb;
                const unsigned short* pa = &Qs[qt16 * 16 + l16][kc * 32 + half * 8];
                fa.h[0] = *reinterpret_cast<const bf16x8*>(pa);
                fa.h[1] = *reinterpret_cast<const bf16x8*>(pa + 16);
                const unsigned short* pb = &Ks[dgrp * 16 + l16][kc * 32 + half * 8];
                fb.h[0] = *reinterpret_cast<const bf16x8*>(pb);
                fb.h[1] = *reinterpret_cast<const bf16x8*>(pb + 16);
                sacc = __builtin_amdgcn_wmma_f32_16x16x32_bf16(
                    false, fa.v, false, fb.v, (short)0, sacc, false, false);
            }
#pragma unroll
            for (int r = 0; r < 8; r++)
                Ss[qt16 * 16 + half * 8 + r][dgrp * 16 + l16] =
                    sacc[r] * 0.08838834764831845f;  // 1/sqrt(128)
        }
        __syncthreads();

        // online softmax, one thread per q row, branch-free via valid count
        if (tid < 64) {
            const int q = tid;
            int nval = row0 + q - k0 + 1;
            nval = nval < 0 ? 0 : (nval > 32 ? 32 : nval);
            float mOld = m_s[q];
            float rmax = -3.0e38f;
            for (int j = 0; j < nval; j++) rmax = fmaxf(rmax, Ss[q][j]);
            float mNew  = fmaxf(mOld, rmax);
            float alpha = __expf(mOld - mNew);
            float rsum  = 0.f;
            for (int j = 0; j < nval; j++) {
                float p = __expf(Ss[q][j] - mNew);
                rsum += p;
                Ps[q][j] = f2bf(p);
            }
            for (int j = nval; j < 32; j++) Ps[q][j] = 0;
            m_s[q]  = mNew;
            l_s[q]  = l_s[q] * alpha + rsum;
            al_s[q] = alpha;
        }
        __syncthreads();

        // O = O*alpha + P @ V  (wave: rows qt16, 4 d-tiles in its 64-col half)
        {
            float av[8];
#pragma unroll
            for (int r = 0; r < 8; r++) av[r] = al_s[qt16 * 16 + half * 8 + r];
            FragAB fa;
            const unsigned short* pa = &Ps[qt16 * 16 + l16][half * 8];
            fa.h[0] = *reinterpret_cast<const bf16x8*>(pa);
            fa.h[1] = *reinterpret_cast<const bf16x8*>(pa + 16);
#pragma unroll
            for (int dt = 0; dt < 4; dt++) {
#pragma unroll
                for (int r = 0; r < 8; r++) oacc[dt][r] *= av[r];
                FragAB fb;
                const unsigned short* pb = &Vts[dgrp * 64 + dt * 16 + l16][half * 8];
                fb.h[0] = *reinterpret_cast<const bf16x8*>(pb);
                fb.h[1] = *reinterpret_cast<const bf16x8*>(pb + 16);
                oacc[dt] = __builtin_amdgcn_wmma_f32_16x16x32_bf16(
                    false, fa.v, false, fb.v, (short)0, oacc[dt], false, false);
            }
        }
        __syncthreads();
    }

    // finalize: divide by softmax denominator, write bf16
    float inv[8];
#pragma unroll
    for (int r = 0; r < 8; r++) {
        float l = l_s[qt16 * 16 + half * 8 + r];
        inv[r] = (l > 0.f) ? 1.0f / l : 0.f;
    }
#pragma unroll
    for (int dt = 0; dt < 4; dt++) {
        int col = h * HD + dgrp * 64 + dt * 16 + l16;
#pragma unroll
        for (int r = 0; r < 8; r++) {
            size_t idx = (size_t)(b * T + row0 + qt16 * 16 + half * 8 + r) * D + col;
            Og[idx] = f2bf(oacc[dt][r] * inv[r]);
        }
    }
}

// ---------------------------------------------------------------------------
// Host launcher
// ---------------------------------------------------------------------------
extern "C" void kernel_launch(void* const* d_in, const int* in_sizes, int n_in,
                              void* d_out, int out_size, void* d_ws, size_t ws_size,
                              hipStream_t stream) {
    const float* x  = (const float*)d_in[0];
    const float* Wq = (const float*)d_in[1];
    const float* Wk = (const float*)d_in[2];
    const float* Wv = (const float*)d_in[3];
    const float* Wo = (const float*)d_in[4];
    const float* bo = (const float*)d_in[5];
    float* out = (float*)d_out;

    const int Bb = 4, T = 2048, D = 2048;
    const int M = Bb * T;  // 8192

    unsigned short* ws  = (unsigned short*)d_ws;
    unsigned short* xbf = ws;                            // M*D (reused as Vt)
    unsigned short* WqT = xbf + (size_t)M * D;           // D*D each
    unsigned short* WkT = WqT + (size_t)D * D;
    unsigned short* WvT = WkT + (size_t)D * D;
    unsigned short* WoT = WvT + (size_t)D * D;
    unsigned short* Qb  = WoT + (size_t)D * D;           // M*D each
    unsigned short* Kb  = Qb + (size_t)M * D;
    unsigned short* Vb  = Kb + (size_t)M * D;
    unsigned short* Ab  = Vb + (size_t)M * D;
    unsigned short* Vtb = xbf;   // x is dead after the QKV GEMMs

    // 1) convert inputs to bf16 (weights transposed -> [n][k])
    size_t nX8 = (size_t)M * D / 8;
    conv_kernel<<<dim3((unsigned)((nX8 + 255) / 256)), 256, 0, stream>>>(x, xbf, nX8);
    dim3 gt(D / 32, D / 32);
    conv_tr_kernel<<<gt, 256, 0, stream>>>(Wq, WqT, D, D);
    conv_tr_kernel<<<gt, 256, 0, stream>>>(Wk, WkT, D, D);
    conv_tr_kernel<<<gt, 256, 0, stream>>>(Wv, WvT, D, D);
    conv_tr_kernel<<<gt, 256, 0, stream>>>(Wo, WoT, D, D);

    // 2) QKV projections (128x256 block tiles)
    dim3 gg(D / 256, M / 128);
    gemm_bf16_kernel<0><<<gg, 256, 0, stream>>>(xbf, WqT, Qb, nullptr, nullptr);
    gemm_bf16_kernel<0><<<gg, 256, 0, stream>>>(xbf, WkT, Kb, nullptr, nullptr);
    gemm_bf16_kernel<0><<<gg, 256, 0, stream>>>(xbf, WvT, Vb, nullptr, nullptr);

    // 2b) transpose V per head: Vt[(b,h,d)][t]  (overwrites xbf, now dead)
    vtr_kernel<<<dim3(M / 32, D / 32), 256, 0, stream>>>(Vb, Vtb);

    // 3) causal flash attention
    attn_kernel<<<dim3(T / 64, 16, Bb), 256, 0, stream>>>(Qb, Kb, Vtb, Ab);

    // 4) output projection + bias (f32 out)
    gemm_bf16_kernel<1><<<gg, 256, 0, stream>>>(Ab, WoT, nullptr, out, bo);
}
